// HyperConvS_12094627906250
// MI455X (gfx1250) — matrix operand order repaired and checked by reading, not compiled, not run on hardware
//
#include <hip/hip_runtime.h>
#include <hip/hip_bf16.h>

// MI455X / gfx1250 implicit-GEMM hyper-conv.
//  - x converted NCHW f32 -> NHWC bf16 (67 MB, fits the 192MB L2)
//  - hypernetwork weight gen writes bf16 weights pre-swizzled into
//    v_wmma_f32_16x16x32_bf16 A-fragment order
//  - conv = GEMM: M=oc(16), N=pixels(16), K=ic*9 (72 x K=32 bf16 WMMA steps)
//  - LDS x-row tiles are double-buffered and filled with
//    GLOBAL_LOAD_ASYNC_TO_LDS_B128 (ASYNCcnt), overlapping fill with WMMA.
// d_ws usage: [0, 67108864) xb bf16 NHWC ; [67108864, +1179648) wb fragments.

typedef __attribute__((ext_vector_type(16))) __bf16 v16bf;
typedef __attribute__((ext_vector_type(8)))  float  v8f;

#define NB 32
#define IC 256
#define OC 256
#define HD 64
#define WD 64
#define BUFH (3 * 66 * 128)   // halfs per LDS x-row buffer (50688 B)

__device__ __forceinline__ unsigned short f2bf(float f) {
  union { float f; unsigned u; } v; v.f = f;
  unsigned u = v.u;
  return (unsigned short)((u + 0x7FFFu + ((u >> 16) & 1u)) >> 16); // RNE
}

__device__ __forceinline__ void async_ld_b128(unsigned lds_byte_off, const void* gaddr) {
  // GLOBAL_LOAD_ASYNC_TO_LDS_B128, GV mode: vdst = per-lane LDS byte address,
  // vaddr = per-lane 64-bit global address. Tracked by ASYNCcnt.
  asm volatile("global_load_async_to_lds_b128 %0, %1, off"
               :: "v"(lds_byte_off), "v"(gaddr) : "memory");
}

__device__ __forceinline__ void wait_async0() {
  asm volatile("s_wait_asynccnt 0x0" ::: "memory");
}

// ---------------------------------------------------------------------------
// Pass 1: x (N,C,H,W) f32  ->  xb (N,H,W,C) bf16, via LDS tile transpose.
// ---------------------------------------------------------------------------
__global__ void xpose_bf16_kernel(const float* __restrict__ x,
                                  unsigned short* __restrict__ xb) {
  __shared__ unsigned short tile[64][72];
  int blk = blockIdx.x;
  int cb = blk & 3;
  int h  = (blk >> 2) & 63;
  int n  = blk >> 8;
  int t  = threadIdx.x;

  int w  = t & 63;
  int cl = t >> 6;
  const float* xp = x + (((size_t)n * IC) * HD + h) * WD + w;
#pragma unroll 4
  for (int i = 0; i < 16; ++i) {
    int c = cb * 64 + i * 4 + cl;
    tile[w][i * 4 + cl] = f2bf(xp[(size_t)c * (HD * WD)]);
  }
  __syncthreads();

  int c2 = t & 63;
  int wq = t >> 6;
  unsigned short* op = xb + (((size_t)n * HD + h) * WD) * IC + cb * 64 + c2;
#pragma unroll 4
  for (int i = 0; i < 16; ++i) {
    int w2 = i * 4 + wq;
    op[(size_t)w2 * IC] = tile[w2][c2];
  }
}

// ---------------------------------------------------------------------------
// Pass 2: hypernetwork weight generation + scatter into WMMA A-fragment order.
//   weight[oc][ic][kh][kw] = K[l=o_blk*16+i_blk][b=ic%16][k=(oc%16)*9+kh*3+kw]
// A-fragment (16-bit A 16x32): lane<16 holds K {0..15}, lane>=16 holds
// K {16..31}, element hh = K%16. wb addr: ((ot*9+kpos)*8 + c)*512 + lane*16 + hh.
// ---------------------------------------------------------------------------
__global__ void hyper_wgen_kernel(const float* __restrict__ z,
                                  const float* __restrict__ W1,
                                  const float* __restrict__ B1,
                                  const float* __restrict__ W2,
                                  const float* __restrict__ B2,
                                  unsigned short* __restrict__ wb) {
  __shared__ float zl[64];
  __shared__ float av[16 * 64];
  int l = blockIdx.x;
  int t = threadIdx.x;
  if (t < 64) zl[t] = z[l * 64 + t];
  __syncthreads();

  for (int e = t; e < 1024; e += 256) {
    int b = e >> 6, d = e & 63;
    const float* w1p = W1 + (b * 64 + d) * 64;
    float s = B1[b * 64 + d];
#pragma unroll 8
    for (int q = 0; q < 64; ++q) s += w1p[q] * zl[q];
    av[e] = s;
  }
  __syncthreads();

  int o_blk = l >> 4, i_blk = l & 15;
  for (int e = t; e < 2304; e += 256) {
    int b = e / 144, k = e % 144;
    const float* w2p = W2 + (b * 144 + k) * 64;
    const float* ap  = av + b * 64;
    float s = B2[b * 144 + k];
#pragma unroll 8
    for (int d = 0; d < 64; ++d) s += w2p[d] * ap[d];

    int b_k  = k / 9;
    int kpos = k % 9;
    int oc = o_blk * 16 + b_k;
    int ic = i_blk * 16 + b;
    int ot = oc >> 4;
    int m  = oc & 15;
    int c  = ic >> 5;
    int kk = ic & 31;
    int hi = kk >> 4;
    int hh = kk & 15;
    int lane = m + 16 * hi;
    wb[((size_t)((ot * 9 + kpos) * 8 + c) << 9) + lane * 16 + hh] = f2bf(s);
  }
}

// ---------------------------------------------------------------------------
// Pass 3: conv as implicit GEMM with bf16 WMMA, f32 accumulate.
// Workgroup (8 waves) computes one output row (n,h): 256 oc x 64 w.
// Wave (wo,wp): oc base wo*64 (4 tiles), w base wp*32 (2 tiles) -> 8 accums.
// x rows h-1..h+1 staged in double-buffered LDS (128 ch per buffer) via
// async-to-LDS loads; buffer ch=1 fill overlaps WMMA on buffer ch=0.
// ---------------------------------------------------------------------------
__device__ __forceinline__ void issue_fill(unsigned short* xrow, int buf,
                                           const unsigned short* xb,
                                           int n, int h, int ch, int t) {
  // interior: 3 rows x 64 w x 128 c = 3072 x 16B chunks (12/thread)
#pragma unroll 4
  for (int i = 0; i < 12; ++i) {
    int chunk = t + i * 256;
    int r   = chunk >> 10;
    int rem = chunk & 1023;
    int ww  = rem >> 4;
    int co  = (rem & 15) << 3;
    int hs  = h - 1 + r;
    if (hs >= 0 && hs < HD) {
      const void* g = xb + ((((size_t)n * HD + hs) * WD + ww) * IC + ch * 128 + co);
      unsigned lds = (unsigned)(size_t)&xrow[buf * BUFH + (r * 66 + ww + 1) * 128 + co];
      async_ld_b128(lds, g);
    }
  }
}

__global__ void __launch_bounds__(256)
hyper_conv_wmma_kernel(const unsigned short* __restrict__ xb,
                       const unsigned short* __restrict__ wb,
                       float* __restrict__ out) {
  extern __shared__ __attribute__((aligned(32))) unsigned short xrow[]; // 2*BUFH

  int h = blockIdx.x;
  int n = blockIdx.y;
  int t = threadIdx.x;

  int lane = t & 31;
  int wave = t >> 5;
  int wo = wave >> 1;
  int wp = wave & 1;
  int lp = lane & 15;
  int hi = lane >> 4;

  const uint4 z4 = make_uint4(0u, 0u, 0u, 0u);
  // Prologue (position-invariant zeros, both buffers):
  // halo columns ww=0 and 65: 2 buf x 3 r x 2 col x 16 chunks = 192
  if (t < 192) {
    int b = t / 96, rem = t % 96;
    int r = rem >> 5, rem2 = rem & 31;
    int ww = (rem2 >> 4) * 65, co = (rem2 & 15) << 3;
    *(uint4*)&xrow[b * BUFH + (r * 66 + ww) * 128 + co] = z4;
  }
  // out-of-range h rows (h=0 or h=63 workgroups only)
#pragma unroll
  for (int r = 0; r < 3; ++r) {
    int hs = h - 1 + r;
    if (hs < 0 || hs >= HD) {
      for (int chunk = t; chunk < 2112; chunk += 256) { // 2 buf x 1056 chunks
        int b = chunk / 1056, cc = chunk % 1056;
        *(uint4*)&xrow[b * BUFH + r * 66 * 128 + cc * 8] = z4;
      }
    }
  }

  issue_fill(xrow, 0, xb, n, h, 0, t);   // async fill buffer 0 (ic 0..127)
  wait_async0();
  __syncthreads();

  v8f acc[4][2] = {};

  for (int ch = 0; ch < 2; ++ch) {
    if (ch == 0) issue_fill(xrow, 1, xb, n, h, 1, t); // overlap with compute

    for (int c2 = 0; c2 < 4; ++c2) {
      int c = ch * 4 + c2;
      for (int kpos = 0; kpos < 9; ++kpos) {
        int kh = kpos / 3, kw = kpos % 3;
        v16bf bfr[2];
#pragma unroll
        for (int j = 0; j < 2; ++j) {
          int ww = wp * 32 + j * 16 + lp + kw;
          bfr[j] = *(const v16bf*)&xrow[ch * BUFH + (kh * 66 + ww) * 128 + (c2 << 5) + (hi << 4)];
        }
#pragma unroll
        for (int i = 0; i < 4; ++i) {
          int ot = wo * 4 + i;
          v16bf afr = *(const v16bf*)(wb + ((size_t)((ot * 9 + kpos) * 8 + c) << 9) + lane * 16);
          acc[i][0] = __builtin_amdgcn_wmma_f32_16x16x32_bf16(
              false, afr, false, bfr[0], (short)0, acc[i][0], false, false);
          acc[i][1] = __builtin_amdgcn_wmma_f32_16x16x32_bf16(
              false, afr, false, bfr[1], (short)0, acc[i][1], false, false);
        }
      }
    }

    if (ch == 0) {          // buffer 1 fill must be complete before reading it
      wait_async0();
      __syncthreads();
    }
  }

  // store: lane = pixel (N), VGPR v = oc row (M); coalesced 64B per half-wave
#pragma unroll
  for (int i = 0; i < 4; ++i) {
#pragma unroll
    for (int j = 0; j < 2; ++j) {
      int w   = wp * 32 + j * 16 + lp;
      int ocb = wo * 64 + i * 16 + hi * 8;
      float* op = out + (((size_t)n * OC + ocb) * HD + h) * WD + w;
#pragma unroll
      for (int v = 0; v < 8; ++v) op[(size_t)v * HD * WD] = acc[i][j][v];
    }
  }
}

// ---------------------------------------------------------------------------
extern "C" void kernel_launch(void* const* d_in, const int* in_sizes, int n_in,
                              void* d_out, int out_size, void* d_ws, size_t ws_size,
                              hipStream_t stream) {
  const float* x  = (const float*)d_in[0];
  const float* z  = (const float*)d_in[1];
  const float* W1 = (const float*)d_in[2];
  const float* B1 = (const float*)d_in[3];
  const float* W2 = (const float*)d_in[4];
  const float* B2 = (const float*)d_in[5];
  float* out = (float*)d_out;

  unsigned short* xb = (unsigned short*)d_ws;                 // 33,554,432 halfs
  unsigned short* wb = xb + (size_t)NB * HD * WD * IC;        // 589,824 halfs

  xpose_bf16_kernel<<<NB * HD * (IC / 64), 256, 0, stream>>>(x, xb);
  hyper_wgen_kernel<<<256, 256, 0, stream>>>(z, W1, B1, W2, B2, wb);
  hyper_conv_wmma_kernel<<<dim3(HD, NB), 256, 2 * BUFH * sizeof(unsigned short), stream>>>(xb, wb, out);
}
